// ConvexUpsample_35570919146183
// MI455X (gfx1250) — matrix operand order repaired and checked by reading, not compiled, not run on hardware
//
#include <hip/hip_runtime.h>

// ---------------------------------------------------------------------------
// ConvexUpsample (RAFT-style), FACTOR=2, B=4, C=64, H=W=128.
// Pipeline: prep(bf16 layouts) -> conv3x3 (WMMA bf16) -> conv1x1 (WMMA bf16)
//           -> softmax + 9-tap convex combine + pixel shuffle (fp32).
// ---------------------------------------------------------------------------

#define NB    4
#define CIN   64
#define HH    128
#define WW    128
#define COUT1 128
#define K1    576            // 9 taps * 64 ch
#define NPIX  (NB * HH * WW) // 65536
#define MCH   48             // 36 mask channels padded to 48

typedef __attribute__((ext_vector_type(16))) __bf16 v16bf;
typedef __attribute__((ext_vector_type(8)))  float  v8f;

union FragBF { uint4 q[2]; v16bf v; };   // 32 bytes: one WMMA bf16 A or B operand
union AccF   { v8f v; float f[8]; };     // 32 bytes: one WMMA f32 C/D operand

__device__ __forceinline__ unsigned short f2bf(float f) {
  unsigned int u = __float_as_uint(f);
  unsigned int r = u + 0x7FFFu + ((u >> 16) & 1u);   // round-to-nearest-even
  return (unsigned short)(r >> 16);
}

// ------------------------- prep kernels ------------------------------------

// h: NCHW f32 -> hb: [b*H*W + y*W + x][c] bf16 (channel-contiguous for B frags)
__global__ __launch_bounds__(256) void prep_h(const float* __restrict__ h,
                                              unsigned short* __restrict__ hb) {
  int idx = blockIdx.x * 256 + threadIdx.x;          // 4,194,304 threads
  int x = idx & 127;
  int y = (idx >> 7) & 127;
  int c = (idx >> 14) & 63;
  int b = idx >> 20;
  size_t dst = ((size_t)((b * HH + y) * WW + x)) * CIN + c;
  hb[dst] = f2bf(h[idx]);
}

// W1: OIHW (128,64,3,3) f32 -> w1b: [co][tap*64 + ci] bf16
__global__ __launch_bounds__(256) void prep_w1(const float* __restrict__ w,
                                               unsigned short* __restrict__ wb) {
  int idx = blockIdx.x * 256 + threadIdx.x;          // 73,728 threads
  int dx = idx % 3;
  int dy = (idx / 3) % 3;
  int ci = (idx / 9) % 64;
  int co = idx / 576;
  wb[(size_t)co * K1 + (dy * 3 + dx) * 64 + ci] = f2bf(w[idx]);
}

// W2: (36,128,1,1) f32 -> w2b: [48][128] bf16, rows 36..47 zero
__global__ __launch_bounds__(256) void prep_w2(const float* __restrict__ w,
                                               unsigned short* __restrict__ wb) {
  int idx = blockIdx.x * 256 + threadIdx.x;          // 6,144 threads
  int k   = idx & 127;
  int row = idx >> 7;
  wb[idx] = (row < 36) ? f2bf(w[row * 128 + k]) : (unsigned short)0;
}

// --------------------- conv3x3 + bias + relu (WMMA) ------------------------
// Block: 256 thr = 8 waves. Tile: 128 Cout x 64 pixels (one row segment).
// Wave w: Cout rows [16w,16w+16), all 4 pixel sub-tiles of 16.
// K loop: 9 taps x 2 chunks of 32 channels -> 18 wmma_f32_16x16x32_bf16 steps.
__global__ __launch_bounds__(256) void conv3x3_wmma(
    const unsigned short* __restrict__ hb,    // [P][64] bf16
    const unsigned short* __restrict__ w1b,   // [128][576] bf16
    const float* __restrict__ b1,             // [128]
    unsigned short* __restrict__ m) {         // [P][128] bf16
  __shared__ unsigned short slab[3 * 66 * CIN];   // 25,344 B halo slab

  int blk = blockIdx.x;                 // 1024 blocks (2 tiles/row * 128 * 4)
  int b   = blk >> 8;
  int rem = blk & 255;
  int y   = rem >> 1;
  int x0  = (rem & 1) * 64;
  int tid = threadIdx.x;

  // Stage halo slab: rows y-1..y+1, cols x0-1..x0+64, 64 ch, as 16B chunks.
  for (int t = tid; t < 1584; t += 256) {
    int chunk = t & 7;                  // 8 x (8 ch) per (row,col)
    int rc  = t >> 3;
    int col = rc % 66;
    int row = rc / 66;
    int yy = y + row - 1;
    int xx = x0 + col - 1;
    uint4 val = make_uint4(0u, 0u, 0u, 0u);
    if (yy >= 0 && yy < HH && xx >= 0 && xx < WW) {
      val = *(const uint4*)(hb + ((size_t)((b * HH + yy) * WW + xx)) * CIN + chunk * 8);
    }
    ((uint4*)slab)[t] = val;
  }
  __syncthreads();

  int wave = tid >> 5;
  int lane = tid & 31;
  int lm   = lane & 15;
  int lh   = lane >> 4;

  AccF acc[4];
#pragma unroll
  for (int n = 0; n < 4; ++n)
#pragma unroll
    for (int r = 0; r < 8; ++r) acc[n].f[r] = 0.0f;

  for (int kk = 0; kk < 18; ++kk) {
    int tap = kk >> 1;
    int dy  = tap / 3;
    int dx  = tap % 3;
    int cib = (kk & 1) * 32;
    // A fragment (16x32 bf16): lane row = 16*wave + lm; K halves split by lh.
    FragBF A;
    const unsigned short* ar =
        w1b + (size_t)(wave * 16 + lm) * K1 + tap * 64 + cib + lh * 16;
    A.q[0] = *(const uint4*)(ar);
    A.q[1] = *(const uint4*)(ar + 8);
#pragma unroll
    for (int n = 0; n < 4; ++n) {
      // B fragment (32x16 bf16) from LDS slab: column = local pixel + dx.
      FragBF Bf;
      int col = n * 16 + lm + dx;
      const unsigned short* bp = slab + ((dy * 66 + col) * CIN + cib + lh * 16);
      Bf.q[0] = *(const uint4*)(bp);
      Bf.q[1] = *(const uint4*)(bp + 8);
      acc[n].v = __builtin_amdgcn_wmma_f32_16x16x32_bf16(
          false, A.v, false, Bf.v, (short)0, acc[n].v, false, false);
    }
  }

  // Epilogue: bias + relu, pack two bf16 per dword, store m[pixel][cout].
  float bias[8];
#pragma unroll
  for (int r = 0; r < 8; ++r) bias[r] = b1[wave * 16 + lh * 8 + r];

#pragma unroll
  for (int n = 0; n < 4; ++n) {
    int px = n * 16 + lm;
    size_t prow = ((size_t)((b * HH + y) * WW + x0 + px)) * COUT1;
    unsigned short* mp = m + prow + wave * 16 + lh * 8;
#pragma unroll
    for (int r = 0; r < 8; r += 2) {
      float v0 = acc[n].f[r] + bias[r];         v0 = v0 > 0.f ? v0 : 0.f;
      float v1 = acc[n].f[r + 1] + bias[r + 1]; v1 = v1 > 0.f ? v1 : 0.f;
      unsigned int pk = (unsigned int)f2bf(v0) | ((unsigned int)f2bf(v1) << 16);
      *(unsigned int*)(mp + r) = pk;
    }
  }
}

// --------------------- conv1x1 mask GEMM (WMMA) ----------------------------
// Block: 8 waves x 16 pixels = 128 pixels. Wave: 48 mask rows (3 M-tiles) x
// 16 pixels, K = 128 in 4 chunks of 32.
__global__ __launch_bounds__(256) void conv1x1_wmma(
    const unsigned short* __restrict__ mB,    // [P][128] bf16
    const unsigned short* __restrict__ w2b,   // [48][128] bf16
    const float* __restrict__ b2,             // [36]
    float* __restrict__ mask) {               // [P][48] f32
  int p0   = blockIdx.x * 128;
  int tid  = threadIdx.x;
  int wave = tid >> 5;
  int lane = tid & 31;
  int lm   = lane & 15;
  int lh   = lane >> 4;
  int p    = p0 + wave * 16 + lm;

  AccF acc[3];
#pragma unroll
  for (int mt = 0; mt < 3; ++mt)
#pragma unroll
    for (int r = 0; r < 8; ++r) acc[mt].f[r] = 0.0f;

#pragma unroll
  for (int kk = 0; kk < 4; ++kk) {
    FragBF Bf;
    const unsigned short* bp = mB + (size_t)p * COUT1 + kk * 32 + lh * 16;
    Bf.q[0] = *(const uint4*)(bp);
    Bf.q[1] = *(const uint4*)(bp + 8);
#pragma unroll
    for (int mt = 0; mt < 3; ++mt) {
      FragBF A;
      const unsigned short* ar =
          w2b + (size_t)(mt * 16 + lm) * COUT1 + kk * 32 + lh * 16;
      A.q[0] = *(const uint4*)(ar);
      A.q[1] = *(const uint4*)(ar + 8);
      acc[mt].v = __builtin_amdgcn_wmma_f32_16x16x32_bf16(
          false, A.v, false, Bf.v, (short)0, acc[mt].v, false, false);
    }
  }

#pragma unroll
  for (int mt = 0; mt < 3; ++mt) {
#pragma unroll
    for (int r = 0; r < 8; ++r) {
      int ch = mt * 16 + lh * 8 + r;
      float v = (ch < 36) ? 0.25f * (acc[mt].f[r] + b2[ch]) : 0.0f;
      mask[(size_t)p * MCH + ch] = v;
    }
  }
}

// ------------- softmax over 9 taps + convex combine + shuffle --------------
// Thread per source pixel. Folds the 8x input scale into the softmax
// normalizer; float2 stores make the stride-2 pixel-shuffle writes coalesced.
__global__ __launch_bounds__(256) void upsample_combine(
    const float* __restrict__ h,        // NCHW f32 (original input)
    const float* __restrict__ mask,     // [P][48]
    float* __restrict__ out) {          // [B][64][256][256]
  int p = blockIdx.x * 256 + threadIdx.x;   // 65536 threads
  int x = p & 127;
  int y = (p >> 7) & 127;
  int b = p >> 14;

  const float* mrow = mask + (size_t)p * MCH;
  float wv[36];
#pragma unroll
  for (int i = 0; i < 36; i += 4) {
    float4 v = *(const float4*)(mrow + i);
    wv[i] = v.x; wv[i + 1] = v.y; wv[i + 2] = v.z; wv[i + 3] = v.w;
  }
  // 4 independent softmaxes over the 9 taps (q = fy*2 + fx).
#pragma unroll
  for (int q = 0; q < 4; ++q) {
    float mx = wv[q];
#pragma unroll
    for (int t = 1; t < 9; ++t) mx = fmaxf(mx, wv[t * 4 + q]);
    float s = 0.0f;
#pragma unroll
    for (int t = 0; t < 9; ++t) {
      float e = __expf(wv[t * 4 + q] - mx);
      wv[t * 4 + q] = e;
      s += e;
    }
    float inv = 8.0f / s;                       // fold the 8*h scale in here
#pragma unroll
    for (int t = 0; t < 9; ++t) wv[t * 4 + q] *= inv;
  }

  const float* hc = h   + (size_t)b * CIN * HH * WW;
  float*       oc = out + (size_t)b * CIN * (2 * HH) * (2 * WW);
  for (int c = 0; c < CIN; ++c) {
    const float* hp = hc + (size_t)c * (HH * WW);
    float hv[9];
#pragma unroll
    for (int t = 0; t < 9; ++t) {
      int yy = y + t / 3 - 1;
      int xx = x + t % 3 - 1;
      hv[t] = (yy >= 0 && yy < HH && xx >= 0 && xx < WW) ? hp[yy * WW + xx] : 0.0f;
    }
    float a0 = 0.f, a1 = 0.f, a2 = 0.f, a3 = 0.f;
#pragma unroll
    for (int t = 0; t < 9; ++t) {
      float hvt = hv[t];
      a0 += wv[t * 4 + 0] * hvt;
      a1 += wv[t * 4 + 1] * hvt;
      a2 += wv[t * 4 + 2] * hvt;
      a3 += wv[t * 4 + 3] * hvt;
    }
    float* op = oc + (size_t)c * (2 * HH) * (2 * WW);
    float2 r0; r0.x = a0; r0.y = a1;
    float2 r1; r1.x = a2; r1.y = a3;
    *(float2*)(op + (size_t)(2 * y + 0) * (2 * WW) + 2 * x) = r0;
    *(float2*)(op + (size_t)(2 * y + 1) * (2 * WW) + 2 * x) = r1;
  }
}

// ---------------------------------------------------------------------------

extern "C" void kernel_launch(void* const* d_in, const int* in_sizes, int n_in,
                              void* d_out, int out_size, void* d_ws, size_t ws_size,
                              hipStream_t stream) {
  (void)in_sizes; (void)n_in; (void)out_size; (void)ws_size;

  const float* h  = (const float*)d_in[0];
  const float* W1 = (const float*)d_in[1];
  const float* b1 = (const float*)d_in[2];
  const float* W2 = (const float*)d_in[3];
  const float* b2 = (const float*)d_in[4];
  float* out = (float*)d_out;

  // Workspace layout (all 256B-aligned sizes), total ~37.9 MB.
  constexpr size_t S_HB = (size_t)NPIX * CIN * 2;    //  8,388,608 : h NHWC bf16
  constexpr size_t S_W1 = (size_t)COUT1 * K1 * 2;    //    147,456 : W1 bf16
  constexpr size_t S_W2 = (size_t)MCH * COUT1 * 2;   //     12,288 : W2 bf16
  constexpr size_t S_M  = (size_t)NPIX * COUT1 * 2;  // 16,777,216 : m bf16

  char* ws = (char*)d_ws;
  unsigned short* hb   = (unsigned short*)(ws);
  unsigned short* w1b  = (unsigned short*)(ws + S_HB);
  unsigned short* w2b  = (unsigned short*)(ws + S_HB + S_W1);
  unsigned short* mbuf = (unsigned short*)(ws + S_HB + S_W1 + S_W2);
  float*          mask = (float*)        (ws + S_HB + S_W1 + S_W2 + S_M);

  prep_h <<<(NPIX * CIN) / 256, 256, 0, stream>>>(h, hb);
  prep_w1<<<(COUT1 * CIN * 9) / 256, 256, 0, stream>>>(W1, w1b);
  prep_w2<<<(MCH * COUT1) / 256, 256, 0, stream>>>(W2, w2b);

  conv3x3_wmma<<<NPIX / 64, 256, 0, stream>>>(hb, w1b, b1, mbuf);
  conv1x1_wmma<<<NPIX / 128, 256, 0, stream>>>(mbuf, w2b, b2, mask);
  upsample_combine<<<NPIX / 256, 256, 0, stream>>>(h, mask, out);
}
